// GraphAttentionLayer_65077344469260
// MI455X (gfx1250) — matrix operand order repaired and checked by reading, not compiled, not run on hardware
//
#include <hip/hip_runtime.h>
#include <hip/hip_bf16.h>
#include <math.h>

// ---------------------------------------------------------------------------
// GAT layer for MI455X (gfx1250, wave32).
//   q/k/v projections + final dense: v_wmma_f32_16x16x32_bf16 (bf16 in, f32 acc)
//   activations pre-converted to bf16 so the GEMM inner loop has NO cvt VALU
//   edge scores: wave-per-edge dot, L2-resident gathers
//   softmax+aggregate: segmented (rows sorted), wave-per-(node,head)
// ---------------------------------------------------------------------------

typedef __attribute__((ext_vector_type(16))) __bf16 v16bf;
typedef __attribute__((ext_vector_type(8)))  __bf16 v8bf;
typedef __attribute__((ext_vector_type(8)))  float  v8f;
typedef __attribute__((ext_vector_type(4)))  float  f32x4;

#define DINx 256   // input / hidden / output feature width
#define Hx   8
#define DHx  32

__device__ __forceinline__ __bf16 f2bf(float f) {
    // round-to-nearest-even fp32 -> bf16
    unsigned u = __builtin_bit_cast(unsigned, f);
    u += 0x7FFFu + ((u >> 16) & 1u);
    unsigned short h = (unsigned short)(u >> 16);
    return __builtin_bit_cast(__bf16, h);
}

// ---------------------------------------------------------------------------
// Kernel 0a: pack Wq/Wk/Wv/Wo into K-major bf16: Wt[mat][c][k], c = output col.
// Wq/Wk/Wv are [H, 256, 32] (col c -> head c>>5, unit c&31); Wo is [256, 256].
// ---------------------------------------------------------------------------
__global__ void pack_weights(const float* __restrict__ Wq, const float* __restrict__ Wk,
                             const float* __restrict__ Wv, const float* __restrict__ Wo,
                             __bf16* __restrict__ wt) {
    int idx = blockIdx.x * 256 + threadIdx.x;      // 4 * 65536 total
    int mat = idx >> 16;
    int rem = idx & 0xFFFF;
    int c = rem >> 8;
    int k = rem & 255;
    float v;
    if (mat == 0)      v = Wq[(c >> 5) * 8192 + k * 32 + (c & 31)];
    else if (mat == 1) v = Wk[(c >> 5) * 8192 + k * 32 + (c & 31)];
    else if (mat == 2) v = Wv[(c >> 5) * 8192 + k * 32 + (c & 31)];
    else               v = Wo[k * 256 + c];
    wt[(size_t)mat * 65536 + (size_t)c * 256 + k] = f2bf(v);
}

// ---------------------------------------------------------------------------
// Kernel 0b: bulk fp32 -> bf16 (row-major preserved). 8 elements per thread.
// ---------------------------------------------------------------------------
__global__ void cvt_f32_bf16(const float* __restrict__ X, __bf16* __restrict__ Xb,
                             long long n) {
    long long i = ((long long)blockIdx.x * 256 + threadIdx.x) * 8;
    if (i >= n) return;
    f32x4 a = *(const f32x4*)(X + i);
    f32x4 b = *(const f32x4*)(X + i + 4);
    v8bf o;
    o[0] = f2bf(a[0]); o[1] = f2bf(a[1]); o[2] = f2bf(a[2]); o[3] = f2bf(a[3]);
    o[4] = f2bf(b[0]); o[5] = f2bf(b[1]); o[6] = f2bf(b[2]); o[7] = f2bf(b[3]);
    *(v8bf*)(Xb + i) = o;
}

// ---------------------------------------------------------------------------
// GEMM: Y[R,256] = act(Xb[R,256](bf16) @ W[256,256] + bias), W K-major bf16.
// One block = 8 waves = one 16-row x 256-col strip. Each wave: 16x32 tile,
// 2 accumulators, K fully unrolled -> 16 v_wmma_f32_16x16x32_bf16.
// Fragment k order: contiguous, k = 16*half + s (wave32). All fragment loads are
// contiguous b128 pairs; no conversion VALU in the loop.
// ---------------------------------------------------------------------------
__global__ void gemm_bf16_256(const __bf16* __restrict__ Xb, const __bf16* __restrict__ Wt,
                              const float* __restrict__ bias, float* __restrict__ Y,
                              int R, int applyLrelu) {
    const int lane = threadIdx.x & 31;
    const int wv   = threadIdx.x >> 5;          // 0..7 -> column tile pair
    const int lnm  = lane & 15;
    const int half = lane >> 4;                 // 0: lanes 0-15, 1: lanes 16-31
    const int rowBase = blockIdx.x * 16;
    int row = rowBase + lnm;
    if (row >= R) row = R - 1;                  // clamp loads; EXEC stays all-ones
    const int cb0 = wv * 32;
    const int cb1 = cb0 + 16;

    v8f acc0 = {};
    v8f acc1 = {};
    const __bf16* xr  = Xb + (size_t)row * DINx;
    const __bf16* wb0 = Wt + (size_t)(cb0 + lnm) * DINx;
    const __bf16* wb1 = Wt + (size_t)(cb1 + lnm) * DINx;

#pragma unroll
    for (int kk = 0; kk < DINx; kk += 32) {
        // A fragment (16x32 bf16): lanes 0-15 row=lnm K=kk..kk+15,
        //                          lanes 16-31 row=lnm K=kk+16..kk+31
        union { v16bf v; v8bf h[2]; } au;
        au.v = *(const v16bf*)(xr + kk + half * 16);

        // B fragment (32x16 bf16): col = cb+lnm, lanes 0-15 K=kk..kk+15,
        // lanes 16-31 K=kk+16..kk+31 -> 16 contiguous bf16 in K-major W.
        v16bf b0 = *(const v16bf*)(wb0 + kk + half * 16);
        v16bf b1 = *(const v16bf*)(wb1 + kk + half * 16);

        acc0 = __builtin_amdgcn_wmma_f32_16x16x32_bf16(false, au.v, false, b0,
                                                       (short)0, acc0, false, false);
        acc1 = __builtin_amdgcn_wmma_f32_16x16x32_bf16(false, au.v, false, b1,
                                                       (short)0, acc1, false, false);
    }

    const float bA = bias[cb0 + lnm];
    const float bB = bias[cb1 + lnm];
#pragma unroll
    for (int j = 0; j < 8; ++j) {
        int orow = rowBase + j + half * 8;      // C/D layout: VGPR j -> M = j + 8*half
        if (orow < R) {
            float v0 = acc0[j] + bA;
            float v1 = acc1[j] + bB;
            if (applyLrelu) {
                v0 = v0 > 0.f ? v0 : 0.2f * v0;
                v1 = v1 > 0.f ? v1 : 0.2f * v1;
            }
            Y[(size_t)orow * DINx + cb0 + lnm] = v0;
            Y[(size_t)orow * DINx + cb1 + lnm] = v1;
        }
    }
}

// ---------------------------------------------------------------------------
// row_ptr via binary search (rows is sorted).
// ---------------------------------------------------------------------------
__global__ void build_rptr(const int* __restrict__ rows, int* __restrict__ rptr,
                           int Nn, int E) {
    int n = blockIdx.x * blockDim.x + threadIdx.x;
    if (n > Nn) return;
    int lo = 0, hi = E;
    while (lo < hi) {
        int mid = (lo + hi) >> 1;
        if (rows[mid] < n) lo = mid + 1; else hi = mid;
    }
    rptr[n] = lo;
}

// ---------------------------------------------------------------------------
// Edge scores: one wave per edge. Lane i covers dims 8i..8i+7; groups of 4
// lanes reduce to one head dot (head h = lane>>2). scores layout [h][e].
// ---------------------------------------------------------------------------
__global__ void edge_scores(const float* __restrict__ Q, const float* __restrict__ Kp,
                            const int* __restrict__ rows, const int* __restrict__ cols,
                            float* __restrict__ scores, int E) {
    int e = blockIdx.x * 8 + (threadIdx.x >> 5);
    if (e >= E) return;
    int lane = threadIdx.x & 31;
    int r = rows[e], c = cols[e];
    const f32x4* q = (const f32x4*)(Q  + (size_t)r * DINx + lane * 8);
    const f32x4* k = (const f32x4*)(Kp + (size_t)c * DINx + lane * 8);
    f32x4 q0 = q[0], q1 = q[1], k0 = k[0], k1 = k[1];
    float p = q0[0]*k0[0] + q0[1]*k0[1] + q0[2]*k0[2] + q0[3]*k0[3]
            + q1[0]*k1[0] + q1[1]*k1[1] + q1[2]*k1[2] + q1[3]*k1[3];
    p += __shfl_xor(p, 1);
    p += __shfl_xor(p, 2);                      // sum over aligned group of 4 lanes
    if ((lane & 3) == 0)
        scores[(size_t)(lane >> 2) * E + e] = p;
}

// ---------------------------------------------------------------------------
// Segmented softmax + SpMM aggregate. One block per node, wave w = head w,
// lane d owns output dim d of that head. Exact 3-phase softmax; result is
// stored bf16 so the final GEMM gets a conversion-free A operand.
// ---------------------------------------------------------------------------
__global__ void softmax_agg(const float* __restrict__ scores, const float* __restrict__ V,
                            const int* __restrict__ cols, const int* __restrict__ rptr,
                            __bf16* __restrict__ AGG, int E) {
    int n = blockIdx.x;
    int h = threadIdx.x >> 5;
    int lane = threadIdx.x & 31;
    int s0 = rptr[n], s1 = rptr[n + 1];
    const float* sh = scores + (size_t)h * E;

    float m = -INFINITY;
    for (int e = s0 + lane; e < s1; e += 32) m = fmaxf(m, sh[e]);
    for (int off = 16; off; off >>= 1) m = fmaxf(m, __shfl_xor(m, off));

    float den = 0.f;
    for (int e = s0 + lane; e < s1; e += 32) den += __expf(sh[e] - m);
    for (int off = 16; off; off >>= 1) den += __shfl_xor(den, off);
    float inv = den > 0.f ? 1.f / den : 0.f;

    float acc = 0.f;
    for (int e = s0; e < s1; ++e) {             // lanes iterate together: sh[e] broadcast
        float w = __expf(sh[e] - m) * inv;
        int c = cols[e];
        acc += w * V[(size_t)c * DINx + h * DHx + lane];   // coalesced 128B per head
    }
    AGG[(size_t)n * DINx + h * DHx + lane] = f2bf(acc);
}

// ---------------------------------------------------------------------------
// Host launch. Workspace layout (~257 MB):
//   [0, 512K)   packed bf16 weights (4 x 65536)
//   qb, mb      bf16 copies of query/memory (25.6 MB each)
//   Q, K, V     fp32 projections (51.2 MB each)
//   AGGb        bf16 aggregate (25.6 MB)
//   scores      H*E fp32 (25.6 MB)
//   row_ptr     (N+1) int
// ---------------------------------------------------------------------------
extern "C" void kernel_launch(void* const* d_in, const int* in_sizes, int n_in,
                              void* d_out, int out_size, void* d_ws, size_t ws_size,
                              hipStream_t stream) {
    const float* query  = (const float*)d_in[0];
    const float* memory = (const float*)d_in[1];
    const int*   rows   = (const int*)d_in[2];
    const int*   cols   = (const int*)d_in[3];
    const float* Wq = (const float*)d_in[4];
    const float* bq = (const float*)d_in[5];
    const float* Wk = (const float*)d_in[6];
    const float* bk = (const float*)d_in[7];
    const float* Wv = (const float*)d_in[8];
    const float* bv = (const float*)d_in[9];
    const float* Wo = (const float*)d_in[10];
    const float* bo = (const float*)d_in[11];
    float* out = (float*)d_out;

    const int N = in_sizes[0] / DINx;   // 50000
    const int M = in_sizes[1] / DINx;   // 50000
    const int E = in_sizes[2];          // 800000

    char* ws = (char*)d_ws;
    __bf16* wt = (__bf16*)ws;
    size_t off = 512 * 1024;
    __bf16* qb = (__bf16*)(ws + off); off += (size_t)N * DINx * 2;
    __bf16* mb = (__bf16*)(ws + off); off += (size_t)M * DINx * 2;
    float* Qb  = (float*)(ws + off); off += (size_t)N * DINx * 4;
    float* Kb  = (float*)(ws + off); off += (size_t)M * DINx * 4;
    float* Vb  = (float*)(ws + off); off += (size_t)M * DINx * 4;
    __bf16* AGGb = (__bf16*)(ws + off); off += (size_t)N * DINx * 2;
    float* scores = (float*)(ws + off); off += (size_t)Hx * E * 4;
    int* rptr  = (int*)(ws + off);

    pack_weights<<<1024, 256, 0, stream>>>(Wq, Wk, Wv, Wo, wt);

    long long nq = (long long)N * DINx, nm = (long long)M * DINx;
    cvt_f32_bf16<<<(int)((nq / 8 + 255) / 256), 256, 0, stream>>>(query,  qb, nq);
    cvt_f32_bf16<<<(int)((nm / 8 + 255) / 256), 256, 0, stream>>>(memory, mb, nm);

    gemm_bf16_256<<<(N + 15) / 16, 256, 0, stream>>>(qb, wt,             bq, Qb, N, 1);
    gemm_bf16_256<<<(M + 15) / 16, 256, 0, stream>>>(mb, wt + 1 * 65536, bk, Kb, M, 1);
    gemm_bf16_256<<<(M + 15) / 16, 256, 0, stream>>>(mb, wt + 2 * 65536, bv, Vb, M, 1);

    build_rptr<<<(N + 1 + 255) / 256, 256, 0, stream>>>(rows, rptr, N, E);
    edge_scores<<<(E + 7) / 8, 256, 0, stream>>>(Qb, Kb, rows, cols, scores, E);
    softmax_agg<<<N, 256, 0, stream>>>(scores, Vb, cols, rptr, AGGb, E);

    gemm_bf16_256<<<(N + 15) / 16, 256, 0, stream>>>(AGGb, wt + 3 * 65536, bo, out, N, 0);
}
